// ProductKeyMemory_49598282334705
// MI455X (gfx1250) — matrix operand order repaired and checked by reading, not compiled, not run on hardware
//
#include <hip/hip_runtime.h>
#include <hip/hip_bf16.h>
#include <cstddef>

// CDNA5 WMMA vector types (wave32)
typedef __attribute__((ext_vector_type(16))) _Float16 v16h;
typedef __attribute__((ext_vector_type(8)))  _Float16 v8h;
typedef __attribute__((ext_vector_type(8)))  float    v8f;

#define TOPK 32
#define D_FULL 1024
#define D_HALF 512
#define N_SMALL 512
#define M_DIM 512
#define TOK_PER_BLK 16

__device__ __forceinline__ v8h cvt8(const float* __restrict__ p) {
    // 8 consecutive f32 -> 8 f16 (two global_load_b128 + v_cvt)
    const float4* q = reinterpret_cast<const float4*>(p);
    float4 a = q[0], b = q[1];
    v8h r;
    r[0] = (_Float16)a.x; r[1] = (_Float16)a.y;
    r[2] = (_Float16)a.z; r[3] = (_Float16)a.w;
    r[4] = (_Float16)b.x; r[5] = (_Float16)b.y;
    r[6] = (_Float16)b.z; r[7] = (_Float16)b.w;
    return r;
}

__global__ __launch_bounds__(256) void ProductKeyMemory_49598282334705_kernel(
    const float* __restrict__ x,       // [8192][1024]
    const float* __restrict__ values,  // [262144][512]
    const float* __restrict__ keys1,   // [512][512]
    const float* __restrict__ keys2,   // [512][512]
    float* __restrict__ y)             // [8192][512]
{
    __shared__ float sScores[TOK_PER_BLK][D_HALF];      // 32 KB, reused per half
    __shared__ float sTopV[2][TOK_PER_BLK][TOPK];       // 4 KB
    __shared__ int   sTopI[2][TOK_PER_BLK][TOPK];       // 4 KB

    const int w      = threadIdx.x >> 5;    // wave id (0..7)
    const int lane   = threadIdx.x & 31;
    const int lhalf  = lane & 15;           // M (for A) / N (for B/C) within 16
    const int tok0   = blockIdx.x * TOK_PER_BLK;
    const float NEG  = -3.402823466e38f;

    // ---------------- Phase 1: per-half score GEMM (WMMA) + exact top-32 ----
    for (int h = 0; h < 2; ++h) {
        const float* __restrict__ keys = (h == 0) ? keys1 : keys2;
        // A row for this lane: token tok0+lhalf, features [h*512, h*512+512)
        const float* __restrict__ xrow =
            x + (size_t)(tok0 + lhalf) * D_FULL + h * D_HALF;
        // 16-bit A layout: lanes 0-15 hold K{0..7,16..23}, lanes 16-31 K{8..15,24..31}
        const int aShift = (lane < 16) ? 0 : 8;
        // 16-bit B layout: lanes 0-15 hold K{0..7,16..23}, lanes 16-31 K{8..15,24..31}
        const int bShift = (lane < 16) ? 0 : 8;

        v8f acc[4] = {};   // wave covers score columns [w*64, w*64+64) as 4 N-tiles
        for (int k0 = 0; k0 < D_HALF; k0 += 32) {
            v16h A;
            {
                const float* pa = xrow + k0 + aShift;
                *reinterpret_cast<v8h*>(&A)       = cvt8(pa);       // K base..base+7
                *(reinterpret_cast<v8h*>(&A) + 1) = cvt8(pa + 16);  // K base+16..+23
            }
            #pragma unroll
            for (int t = 0; t < 4; ++t) {
                const int bcol = w * 64 + t * 16 + lhalf;           // key index b
                const float* pb = keys + (size_t)bcol * D_HALF + k0 + bShift;
                v16h B;
                *reinterpret_cast<v8h*>(&B)       = cvt8(pb);
                *(reinterpret_cast<v8h*>(&B) + 1) = cvt8(pb + 16);
                acc[t] = __builtin_amdgcn_wmma_f32_16x16x32_f16(
                    false, A, false, B, (short)0, acc[t], false, false);
            }
        }
        // C layout: VGPR r, lanes 0-15 -> M=r, lanes 16-31 -> M=r+8; N = lhalf
        {
            const int rbase = (lane < 16) ? 0 : 8;
            #pragma unroll
            for (int t = 0; t < 4; ++t) {
                const int col = w * 64 + t * 16 + lhalf;
                #pragma unroll
                for (int r = 0; r < 8; ++r)
                    sScores[rbase + r][col] = acc[t][r];
            }
        }
        __syncthreads();

        // exact top-32 of 512 per token; wave w handles tokens w and w+8
        for (int token = w; token < TOK_PER_BLK; token += 8) {
            float ls[16];
            #pragma unroll
            for (int j = 0; j < 16; ++j) ls[j] = sScores[token][lane * 16 + j];
            for (int r = 0; r < TOPK; ++r) {
                float bv = NEG; int bj = 0;
                #pragma unroll
                for (int j = 0; j < 16; ++j)
                    if (ls[j] > bv) { bv = ls[j]; bj = j; }
                int bidx = lane * 16 + bj;
                #pragma unroll
                for (int off = 16; off >= 1; off >>= 1) {
                    float ov = __shfl_xor(bv, off, 32);
                    int   oi = __shfl_xor(bidx, off, 32);
                    if (ov > bv || (ov == bv && oi < bidx)) { bv = ov; bidx = oi; }
                }
                if (lane == 0) { sTopV[h][token][r] = bv; sTopI[h][token][r] = bidx; }
                if ((bidx >> 4) == lane) {
                    #pragma unroll
                    for (int j = 0; j < 16; ++j)
                        if (j == (bidx & 15)) ls[j] = NEG;
                }
            }
        }
        __syncthreads();
    }

    // ------------- Phase 2: product top-32, softmax, gather, output ---------
    for (int token = w; token < TOK_PER_BLK; token += 8) {
        // lane i owns row i of the 32x32 product matrix
        const float s1 = sTopV[0][token][lane];
        float p[32];
        #pragma unroll
        for (int j = 0; j < 32; ++j) p[j] = s1 * sTopV[1][token][j];

        float myv = 0.0f; int myidx = 0;    // round-r winner parked in lane r
        for (int r = 0; r < TOPK; ++r) {
            float bv = NEG; int bj = 0;
            #pragma unroll
            for (int j = 0; j < 32; ++j)
                if (p[j] > bv) { bv = p[j]; bj = j; }
            int code = lane * 32 + bj;      // flat index i*32+j (tiebreak = lowest)
            #pragma unroll
            for (int off = 16; off >= 1; off >>= 1) {
                float ov = __shfl_xor(bv, off, 32);
                int   oc = __shfl_xor(code, off, 32);
                if (ov > bv || (ov == bv && oc < code)) { bv = ov; code = oc; }
            }
            if (lane == r) {
                myv = bv;
                myidx = sTopI[0][token][code >> 5] * N_SMALL
                      + sTopI[1][token][code & 31];
            }
            if ((code >> 5) == lane) {
                #pragma unroll
                for (int j = 0; j < 32; ++j)
                    if (j == (code & 31)) p[j] = NEG;
            }
        }

        // softmax over 32 winners (entry 0 is the max: selection was descending)
        float m = __shfl(myv, 0, 32);
        float e = __expf(myv - m);
        float sum = e;
        #pragma unroll
        for (int off = 16; off >= 1; off >>= 1) sum += __shfl_xor(sum, off, 32);
        const float wgt = e / sum;

        // gather: lane l accumulates columns [l*16, l*16+16) of the 512-wide rows
        float4 a0 = {0,0,0,0}, a1 = {0,0,0,0}, a2 = {0,0,0,0}, a3 = {0,0,0,0};
        for (int ei = 0; ei < TOPK; ++ei) {
            const float we = __shfl(wgt, ei, 32);
            const int   vi = __shfl(myidx, ei, 32);
            const float4* vp =
                reinterpret_cast<const float4*>(values + (size_t)vi * M_DIM) + lane * 4;
            // prefetch next row (global_prefetch_b8)
            const int ni = __shfl(myidx, (ei < TOPK - 1) ? ei + 1 : ei, 32);
            __builtin_prefetch(values + (size_t)ni * M_DIM + lane * 16, 0, 1);
            float4 v0 = vp[0], v1 = vp[1], v2 = vp[2], v3 = vp[3];
            a0.x = fmaf(we, v0.x, a0.x); a0.y = fmaf(we, v0.y, a0.y);
            a0.z = fmaf(we, v0.z, a0.z); a0.w = fmaf(we, v0.w, a0.w);
            a1.x = fmaf(we, v1.x, a1.x); a1.y = fmaf(we, v1.y, a1.y);
            a1.z = fmaf(we, v1.z, a1.z); a1.w = fmaf(we, v1.w, a1.w);
            a2.x = fmaf(we, v2.x, a2.x); a2.y = fmaf(we, v2.y, a2.y);
            a2.z = fmaf(we, v2.z, a2.z); a2.w = fmaf(we, v2.w, a2.w);
            a3.x = fmaf(we, v3.x, a3.x); a3.y = fmaf(we, v3.y, a3.y);
            a3.z = fmaf(we, v3.z, a3.z); a3.w = fmaf(we, v3.w, a3.w);
        }
        float4* op =
            reinterpret_cast<float4*>(y + (size_t)(tok0 + token) * M_DIM) + lane * 4;
        op[0] = a0; op[1] = a1; op[2] = a2; op[3] = a3;
    }
}

extern "C" void kernel_launch(void* const* d_in, const int* in_sizes, int n_in,
                              void* d_out, int out_size, void* d_ws, size_t ws_size,
                              hipStream_t stream) {
    const float* x      = (const float*)d_in[0];   // (8,1024,1024)
    const float* values = (const float*)d_in[1];   // (262144,512)
    const float* keys1  = (const float*)d_in[2];   // (512,512)
    const float* keys2  = (const float*)d_in[3];   // (512,512)
    float* y = (float*)d_out;                      // (8,1024,512)

    const int tokens = in_sizes[0] / D_FULL;       // 8192
    const int blocks = tokens / TOK_PER_BLK;       // 512
    hipLaunchKernelGGL(ProductKeyMemory_49598282334705_kernel,
                       dim3(blocks), dim3(256), 0, stream,
                       x, values, keys1, keys2, y);
}
